// TransGAT_10866267259407
// MI455X (gfx1250) — matrix operand order repaired and checked by reading, not compiled, not run on hardware
//
#include <hip/hip_runtime.h>
#include <hip/hip_bf16.h>
#include <cstdint>

typedef __attribute__((ext_vector_type(16))) __bf16 v16bf;
typedef __attribute__((ext_vector_type(8)))  __bf16 v8bf;
typedef __attribute__((ext_vector_type(8)))  float  v8f;

#define LRELU_ALPHA 0.2f
#define LDS_STRIDE 48   // bf16 elems per LDS row: 96B, keeps 16B-aligned fragment loads

__device__ __forceinline__ __bf16 f2bf(float f) {
  union { float f; uint32_t u; } in; in.f = f;
  uint32_t u = in.u;
  uint32_t r = (u + 0x7FFFu + ((u >> 16) & 1u)) >> 16;   // round-to-nearest-even
  union { unsigned short s; __bf16 b; } out; out.s = (unsigned short)r;
  return out.b;
}

__device__ __forceinline__ v16bf load_frag(const __bf16* p0, const __bf16* p1) {
  v8bf lo = *reinterpret_cast<const v8bf*>(p0);
  v8bf hi = *reinterpret_cast<const v8bf*>(p1);
  v16bf r;
#pragma unroll
  for (int i = 0; i < 8; ++i) { r[i] = lo[i]; r[8 + i] = hi[i]; }
  return r;
}

// ---------------------------------------------------------------------------
// neighbor = (adj @ x) / max(rowsum(adj), 1e-12)
// Block: 128 rows x full N=256. adj read exactly once; degree fused into the
// A-tile load via LDS float atomics. 8 waves, each wave: 16 rows x 16 tiles.
// ---------------------------------------------------------------------------
__global__ __launch_bounds__(256) void k_adj_gemm(const float* __restrict__ adj,
                                                  const float* __restrict__ x,
                                                  float* __restrict__ nb,
                                                  int Nn, int F /*==256*/) {
  __shared__ __align__(16) __bf16 As[128 * LDS_STRIDE];
  __shared__ __align__(16) __bf16 BsT[256 * LDS_STRIDE];
  __shared__ float rowAcc[128];

  const int tid  = threadIdx.x;
  const int wave = tid >> 5, lane = tid & 31;
  const int half = lane >> 4, ln = lane & 15;
  const int bm   = blockIdx.x * 128;

  if (tid < 128) rowAcc[tid] = 0.f;
  __syncthreads();

  v8f acc[16];
#pragma unroll
  for (int t = 0; t < 16; ++t)
#pragma unroll
    for (int i = 0; i < 8; ++i) acc[t][i] = 0.f;

  for (int k0 = 0; k0 < Nn; k0 += 32) {
    { // A tile (adj) 128x32 -> bf16 LDS, fused row-sum
      const int r  = tid >> 1;
      const int cb = (tid & 1) * 16;
      const float4* ap = reinterpret_cast<const float4*>(adj + (size_t)(bm + r) * Nn + k0 + cb);
      float rs = 0.f;
#pragma unroll
      for (int j = 0; j < 4; ++j) {
        float4 v = ap[j];
        rs += v.x + v.y + v.z + v.w;
        __bf16* d = &As[r * LDS_STRIDE + cb + j * 4];
        d[0] = f2bf(v.x); d[1] = f2bf(v.y); d[2] = f2bf(v.z); d[3] = f2bf(v.w);
      }
      atomicAdd(&rowAcc[r], rs);
    }
    { // B tile (x) 32x256 -> transposed bf16 LDS
      const int kr = tid >> 3;
      const int nc = (tid & 7) * 32;
      const float4* bp = reinterpret_cast<const float4*>(x + (size_t)(k0 + kr) * F + nc);
#pragma unroll
      for (int j = 0; j < 8; ++j) {
        float4 v = bp[j];
        const int c = nc + j * 4;
        BsT[(c + 0) * LDS_STRIDE + kr] = f2bf(v.x);
        BsT[(c + 1) * LDS_STRIDE + kr] = f2bf(v.y);
        BsT[(c + 2) * LDS_STRIDE + kr] = f2bf(v.z);
        BsT[(c + 3) * LDS_STRIDE + kr] = f2bf(v.w);
      }
    }
    __syncthreads();

    const int arow = wave * 16 + ln;
    v16bf af = load_frag(&As[arow * LDS_STRIDE + half * 8],
                         &As[arow * LDS_STRIDE + 16 + half * 8]);
#pragma unroll
    for (int ct = 0; ct < 16; ++ct) {
      const int col = ct * 16 + ln;
      v16bf bfr = load_frag(&BsT[col * LDS_STRIDE + half * 8],
                            &BsT[col * LDS_STRIDE + 16 + half * 8]);
      acc[ct] = __builtin_amdgcn_wmma_f32_16x16x32_bf16(false, af, false, bfr,
                                                        (short)0, acc[ct], false, false);
    }
    __syncthreads();
  }

  // epilogue: divide by degree, store neighbor
#pragma unroll
  for (int ct = 0; ct < 16; ++ct)
#pragma unroll
    for (int v = 0; v < 8; ++v) {
      const int rl  = wave * 16 + v + 8 * half;
      const float dg = fmaxf(rowAcc[rl], 1e-12f);
      nb[(size_t)(bm + rl) * F + ct * 16 + ln] = acc[ct][v] / dg;
    }
}

// ---------------------------------------------------------------------------
// Generic WMMA GEMM with optional concatenated-K:
// C[M x Nc] = [A0 | A1] @ [B0 ; B1], A0:[M x K1], A1:[M x K2], Bi:[Ki x Nc].
// Block tile 128x64, 8 waves, each wave 16 rows x 4 tiles.
// ---------------------------------------------------------------------------
__global__ __launch_bounds__(256) void k_gemm_cat(const float* __restrict__ A0,
                                                  const float* __restrict__ A1,
                                                  const float* __restrict__ B0,
                                                  const float* __restrict__ B1,
                                                  float* __restrict__ C,
                                                  int M, int Nc, int K1, int K2) {
  __shared__ __align__(16) __bf16 As[128 * LDS_STRIDE];
  __shared__ __align__(16) __bf16 BsT[64 * LDS_STRIDE];

  const int tid  = threadIdx.x;
  const int wave = tid >> 5, lane = tid & 31;
  const int half = lane >> 4, ln = lane & 15;
  const int bm   = blockIdx.x * 128;
  const int bn   = blockIdx.y * 64;

  v8f acc[4];
#pragma unroll
  for (int t = 0; t < 4; ++t)
#pragma unroll
    for (int i = 0; i < 8; ++i) acc[t][i] = 0.f;

  const int Ktot = K1 + K2;
  for (int k0 = 0; k0 < Ktot; k0 += 32) {
    const bool seg1 = (k0 >= K1);
    const float* A = seg1 ? A1 : A0;
    const float* B = seg1 ? B1 : B0;
    const int kk  = seg1 ? (k0 - K1) : k0;
    const int ldA = seg1 ? K2 : K1;
    { // A tile 128x32
      const int r  = tid >> 1;
      const int cb = (tid & 1) * 16;
      const float4* ap = reinterpret_cast<const float4*>(A + (size_t)(bm + r) * ldA + kk + cb);
#pragma unroll
      for (int j = 0; j < 4; ++j) {
        float4 v = ap[j];
        __bf16* d = &As[r * LDS_STRIDE + cb + j * 4];
        d[0] = f2bf(v.x); d[1] = f2bf(v.y); d[2] = f2bf(v.z); d[3] = f2bf(v.w);
      }
    }
    { // B tile 32x64, transposed into LDS
      const int kr = tid >> 3;
      const int nc = (tid & 7) * 8;
      const float4* bp = reinterpret_cast<const float4*>(B + (size_t)(kk + kr) * Nc + bn + nc);
#pragma unroll
      for (int j = 0; j < 2; ++j) {
        float4 v = bp[j];
        const int c = nc + j * 4;
        BsT[(c + 0) * LDS_STRIDE + kr] = f2bf(v.x);
        BsT[(c + 1) * LDS_STRIDE + kr] = f2bf(v.y);
        BsT[(c + 2) * LDS_STRIDE + kr] = f2bf(v.z);
        BsT[(c + 3) * LDS_STRIDE + kr] = f2bf(v.w);
      }
    }
    __syncthreads();

    const int arow = wave * 16 + ln;
    v16bf af = load_frag(&As[arow * LDS_STRIDE + half * 8],
                         &As[arow * LDS_STRIDE + 16 + half * 8]);
#pragma unroll
    for (int ct = 0; ct < 4; ++ct) {
      const int col = ct * 16 + ln;
      v16bf bfr = load_frag(&BsT[col * LDS_STRIDE + half * 8],
                            &BsT[col * LDS_STRIDE + 16 + half * 8]);
      acc[ct] = __builtin_amdgcn_wmma_f32_16x16x32_bf16(false, af, false, bfr,
                                                        (short)0, acc[ct], false, false);
    }
    __syncthreads();
  }

#pragma unroll
  for (int ct = 0; ct < 4; ++ct)
#pragma unroll
    for (int v = 0; v < 8; ++v) {
      const int rl = wave * 16 + v + 8 * half;
      C[(size_t)(bm + rl) * Nc + bn + ct * 16 + ln] = acc[ct][v];
    }
}

// ---------------------------------------------------------------------------
// output = x + (lrelu(G)+1)*r + lrelu(Bt) - nb
// ---------------------------------------------------------------------------
__global__ __launch_bounds__(256) void k_relation(const float* __restrict__ x,
                                                  const float* __restrict__ nb,
                                                  const float* __restrict__ G,
                                                  const float* __restrict__ Bt,
                                                  const float* __restrict__ r,
                                                  float* __restrict__ out,
                                                  int total, int F) {
  int i = blockIdx.x * 256 + threadIdx.x;
  if (i >= total) return;
  const int f = i % F;
  float g = G[i];  g = (g > 0.f ? g : LRELU_ALPHA * g) + 1.f;
  float b = Bt[i]; b = (b > 0.f ? b : LRELU_ALPHA * b);
  out[i] = x[i] + g * r[f] + b - nb[i];
}

// ---------------------------------------------------------------------------
// Per-node attention scores: s1[h,n] = h[h,n,:]·a[h,:D], s2 = h2·a[h,D:]
// One wave32 per (head, node), shuffle reduction.
// ---------------------------------------------------------------------------
__global__ __launch_bounds__(256) void k_scores(const float* __restrict__ h,
                                                const float* __restrict__ h2,
                                                const float* __restrict__ a,
                                                float* __restrict__ s1,
                                                float* __restrict__ s2,
                                                int Nn, int D, int H) {
  const int wave = threadIdx.x >> 5, lane = threadIdx.x & 31;
  const int g = blockIdx.x * 8 + wave;
  if (g >= Nn * H) return;
  const int head = g / Nn, n = g - head * Nn;
  const float* hv  = h  + ((size_t)head * Nn + n) * D;
  const float* h2v = h2 + ((size_t)head * Nn + n) * D;
  const float* av  = a + (size_t)head * 2 * D;
  float r1 = 0.f, r2 = 0.f;
  for (int c = lane; c < D; c += 32) { r1 += hv[c] * av[c]; r2 += h2v[c] * av[D + c]; }
#pragma unroll
  for (int o = 16; o > 0; o >>= 1) { r1 += __shfl_xor(r1, o, 32); r2 += __shfl_xor(r2, o, 32); }
  if (lane == 0) { s1[g] = r1; s2[g] = r2; }
}

__global__ __launch_bounds__(256) void k_zero(float* __restrict__ p, size_t n) {
  size_t i = (size_t)blockIdx.x * 256 + threadIdx.x;
  if (i < n) p[i] = 0.f;
}

// ---------------------------------------------------------------------------
// Edge scatter: w = exp(-lrelu(s1[src]+s2[dst]));
// rowsum[src] += w; h_prime[src,:] += w * h[dst,:]. One wave32 per edge.
// ---------------------------------------------------------------------------
__global__ __launch_bounds__(256) void k_edges(const int* __restrict__ src,
                                               const int* __restrict__ dst,
                                               const float* __restrict__ s1,
                                               const float* __restrict__ s2,
                                               const float* __restrict__ h,
                                               float* __restrict__ hp,
                                               float* __restrict__ rowsum,
                                               int E, int Nn, int D, int H) {
  const int g = (int)((blockIdx.x * 256 + threadIdx.x) >> 5);
  const int lane = threadIdx.x & 31;
  if (g >= E) return;
  const int s = src[g], d = dst[g];
  for (int head = 0; head < H; ++head) {
    const float logit = s1[head * Nn + s] + s2[head * Nn + d];
    const float l = logit > 0.f ? logit : LRELU_ALPHA * logit;
    const float w = __expf(-l);
    if (lane == 0) atomicAdd(&rowsum[head * Nn + s], w);
    const float* hv = h + ((size_t)head * Nn + d) * D;
    float* hps = hp + ((size_t)head * Nn + s) * D;
    for (int c = lane; c < D; c += 32) atomicAdd(&hps[c], w * hv[c]);
  }
}

// ---------------------------------------------------------------------------
// h_k[n, head*D + c] = elu(h_prime / rowsum)
// ---------------------------------------------------------------------------
__global__ __launch_bounds__(256) void k_final(const float* __restrict__ hp,
                                               const float* __restrict__ rowsum,
                                               float* __restrict__ hk,
                                               int Nn, int D, int H) {
  const int idx = blockIdx.x * 256 + threadIdx.x;
  const int total = Nn * H * D;
  if (idx >= total) return;
  const int n = idx / (H * D);
  const int rem = idx - n * (H * D);
  const int head = rem / D;
  const int c = rem - head * D;
  const float v = hp[((size_t)head * Nn + n) * D + c] / rowsum[head * Nn + n];
  hk[idx] = v > 0.f ? v : (__expf(v) - 1.f);
}

// ---------------------------------------------------------------------------
extern "C" void kernel_launch(void* const* d_in, const int* in_sizes, int n_in,
                              void* d_out, int out_size, void* d_ws, size_t ws_size,
                              hipStream_t stream) {
  (void)n_in; (void)out_size; (void)ws_size;
  const float* x   = (const float*)d_in[0];
  const float* adj = (const float*)d_in[1];
  const int*   ei  = (const int*)d_in[2];
  const float* g1  = (const float*)d_in[3];
  const float* g2  = (const float*)d_in[4];
  const float* b1  = (const float*)d_in[5];
  const float* b2  = (const float*)d_in[6];
  const float* r   = (const float*)d_in[7];
  const float* W   = (const float*)d_in[8];
  const float* a   = (const float*)d_in[9];

  const int F = 256, D = 128, H = 3;
  const int Nn = in_sizes[0] / F;       // 8192
  const int E  = in_sizes[2] / 2;       // 270336

  float* ws = (float*)d_ws;
  size_t off = 0;
  float* nb  = ws + off; off += (size_t)Nn * F;
  float* G   = ws + off; off += (size_t)Nn * F;
  float* Bt  = ws + off; off += (size_t)Nn * F;
  float* h   = ws + off; off += (size_t)H * Nn * D;
  float* h2  = ws + off; off += (size_t)H * Nn * D;
  float* s1  = ws + off; off += (size_t)H * Nn;
  float* s2  = ws + off; off += (size_t)H * Nn;
  float* rowsum = ws + off; off += (size_t)H * Nn;      // rowsum + hp contiguous
  float* hp  = ws + off; off += (size_t)H * Nn * D;

  float* hk   = (float*)d_out;                          // [Nn, H*D]
  float* outp = (float*)d_out + (size_t)Nn * H * D;     // [Nn, F]

  // 1) neighbor = normalize(adj) @ x  (degree fused)
  k_adj_gemm<<<Nn / 128, 256, 0, stream>>>(adj, x, nb, Nn, F);

  // 2) G = x@g1 + nb@g2 ; Bt = x@b1 + nb@b2 ; relation epilogue -> output
  k_gemm_cat<<<dim3(Nn / 128, F / 64), 256, 0, stream>>>(x, nb, g1, g2, G, Nn, F, F, F);
  k_gemm_cat<<<dim3(Nn / 128, F / 64), 256, 0, stream>>>(x, nb, b1, b2, Bt, Nn, F, F, F);
  k_relation<<<(Nn * F + 255) / 256, 256, 0, stream>>>(x, nb, G, Bt, r, outp, Nn * F, F);

  // 3) per-head projections
  for (int head = 0; head < H; ++head) {
    const float* Wh = W + (size_t)head * F * D;
    k_gemm_cat<<<dim3(Nn / 128, D / 64), 256, 0, stream>>>(x,    nullptr, Wh, nullptr,
                                                           h  + (size_t)head * Nn * D, Nn, D, F, 0);
    k_gemm_cat<<<dim3(Nn / 128, D / 64), 256, 0, stream>>>(outp, nullptr, Wh, nullptr,
                                                           h2 + (size_t)head * Nn * D, Nn, D, F, 0);
  }

  // 4) scores, zero accumulators, edge scatter, ELU epilogue
  k_scores<<<(Nn * H + 7) / 8, 256, 0, stream>>>(h, h2, a, s1, s2, Nn, D, H);
  const size_t zcount = (size_t)H * Nn * (D + 1);
  k_zero<<<(unsigned)((zcount + 255) / 256), 256, 0, stream>>>(rowsum, zcount);
  k_edges<<<(E + 7) / 8, 256, 0, stream>>>(ei, ei + E, s1, s2, h, hp, rowsum, E, Nn, D, H);
  k_final<<<(Nn * H * D + 255) / 256, 256, 0, stream>>>(hp, rowsum, hk, Nn, D, H);
}
